// WithinSubjectTripletLoss_50663434223746
// MI455X (gfx1250) — matrix operand order repaired and checked by reading, not compiled, not run on hardware
//
#include <hip/hip_runtime.h>

#define NROWS  8192
#define DIM    256
#define MARGIN 1.0f
#define EPSV   1e-6f
#define BIGV   1e30f
#define NT     (NROWS / 16)      // 512 column tiles
#define BPAD   8                 // pad (ushorts) -> row stride 264*2=528B, bank-conflict-free
#define BSTRIDE (DIM + BPAD)

typedef __bf16 bf16x8  __attribute__((ext_vector_type(8)));
typedef __bf16 bf16x16 __attribute__((ext_vector_type(16)));
typedef float  f32x8   __attribute__((ext_vector_type(8)));

__device__ __forceinline__ unsigned short f32_to_bf16(float f) {
  unsigned int u = __float_as_uint(f);
  u += 0x7fffu + ((u >> 16) & 1u);   // round-to-nearest-even
  return (unsigned short)(u >> 16);
}

// ---------------------------------------------------------------------------
// Kernel 0: zero the accumulators (sum, count)
// ---------------------------------------------------------------------------
__global__ void WST_init_kernel(float* __restrict__ sumcnt) {
  if (threadIdx.x < 2) sumcnt[threadIdx.x] = 0.0f;
}

// ---------------------------------------------------------------------------
// Kernel 1: fp32 -> bf16 conversion + per-row squared norms.
// One wave32 per row; each lane handles 8 contiguous dims.
// ---------------------------------------------------------------------------
__global__ __launch_bounds__(256) void WST_prep_kernel(
    const float* __restrict__ emb,
    unsigned short* __restrict__ embh,
    float* __restrict__ sq) {
  int wid  = (int)((blockIdx.x * blockDim.x + threadIdx.x) >> 5);  // row
  int lane = (int)(threadIdx.x & 31);
  if (wid >= NROWS) return;

  const float* src = emb + (size_t)wid * DIM + lane * 8;
  float4 f0 = ((const float4*)src)[0];
  float4 f1 = ((const float4*)src)[1];
  float acc = f0.x*f0.x + f0.y*f0.y + f0.z*f0.z + f0.w*f0.w
            + f1.x*f1.x + f1.y*f1.y + f1.z*f1.z + f1.w*f1.w;

  union { unsigned short s[8]; uint4 v; } pk;
  pk.s[0] = f32_to_bf16(f0.x); pk.s[1] = f32_to_bf16(f0.y);
  pk.s[2] = f32_to_bf16(f0.z); pk.s[3] = f32_to_bf16(f0.w);
  pk.s[4] = f32_to_bf16(f1.x); pk.s[5] = f32_to_bf16(f1.y);
  pk.s[6] = f32_to_bf16(f1.z); pk.s[7] = f32_to_bf16(f1.w);
  *(uint4*)(embh + (size_t)wid * DIM + lane * 8) = pk.v;

  #pragma unroll
  for (int m = 16; m >= 1; m >>= 1) acc += __shfl_xor(acc, m);
  if (lane == 0) sq[wid] = acc;
}

// ---------------------------------------------------------------------------
// Kernel 2: fused Gram-matrix (bf16 WMMA) + hardest-positive / hardest-negative
// mining, with workgroup-level B reuse through LDS (double-buffered).
//
// 8 waves/block, each owning a 16-row A block (held in registers for the whole
// sweep). The 16x256 B tile (8 KB) for each of the 512 column tiles is staged
// into LDS once per workgroup and consumed by all 8 waves -> 8x less L2 read
// traffic than register-only streaming (AI: 16.8 -> 134 FLOP per L2 byte).
//
// VGPR layouts per CDNA5 ISA 7.12.2 (wave32):
//   A 16x32 bf16 : lane = half*16 + m16; lo 8 elems K = kc*32 + half*8,
//                  hi 8 elems K = kc*32 + half*8 + 16
//   B 32x16 bf16 : lane owns column n = l15; K split as A (kc*32+half*8, +16)
//   C 16x16 f32  : element r <-> (M = half*8 + r, N = l15)
// ---------------------------------------------------------------------------
__global__ __launch_bounds__(256) void WST_mine_kernel(
    const unsigned short* __restrict__ embh,
    const float* __restrict__ sq,
    const int* __restrict__ labels,
    const int* __restrict__ sbj,
    int* __restrict__ hp,
    int* __restrict__ hn) {
  __shared__ unsigned short Bt[2][16][BSTRIDE];   // ~16.5 KB of 320 KB WGP LDS

  const int tid  = (int)threadIdx.x;
  const int lane = tid & 31;
  const int wv   = tid >> 5;
  const int rowblk = (int)blockIdx.x * 8 + wv;     // 512 row blocks total
  const int row0   = rowblk * 16;
  const int half = lane >> 4;        // 0: lanes 0-15, 1: lanes 16-31
  const int l15  = lane & 15;

  // Cooperative staging: thread t copies 32B chunk (row = t>>4, chunk = t&15)
  const int srow   = tid >> 4;
  const int schunk = tid & 15;

  // ---- Load A fragments for this 16-row block (held across whole sweep) ----
  bf16x16 a[8];
  {
    const unsigned short* abase = embh + (size_t)(row0 + l15) * DIM;
    #pragma unroll
    for (int kc = 0; kc < 8; ++kc) {
      int kb = kc * 32 + half * 8;
      bf16x8 lo = *(const bf16x8*)(abase + kb);
      bf16x8 hi = *(const bf16x8*)(abase + kb + 16);
      #pragma unroll
      for (int e = 0; e < 8; ++e) { a[kc][e] = lo[e]; a[kc][e + 8] = hi[e]; }
    }
  }

  // ---- Per-lane row metadata (this lane's 8 rows: M = half*8 + r) ----
  int   myLab[8], mySbj[8];
  float mySq[8];
  #pragma unroll
  for (int r = 0; r < 8; ++r) {
    int mrow = row0 + half * 8 + r;
    myLab[r] = labels[mrow];
    mySbj[r] = sbj[mrow];
    mySq[r]  = sq[mrow];
  }

  float bestP[8], bestN[8];
  int   bpi[8], bni[8];
  #pragma unroll
  for (int r = 0; r < 8; ++r) {
    bestP[r] = -BIGV; bestN[r] = BIGV; bpi[r] = -1; bni[r] = -1;
  }

  // ---- Prologue: stage tile 0 into buffer 0 ----
  {
    const uint4* src = (const uint4*)(embh + (size_t)srow * DIM + schunk * 16);
    uint4* dst = (uint4*)&Bt[0][srow][schunk * 16];
    dst[0] = src[0];
    dst[1] = src[1];
  }

  // ---- Sweep all column tiles ----
  for (int jt = 0; jt < NT; ++jt) {
    const int buf = jt & 1;
    __syncthreads();   // tile jt visible; buffer buf^1 free for refill

    // Stage tile jt+1 into the other buffer; prefetch tile jt+2 toward WGP.
    if (jt + 1 < NT) {
      const uint4* src =
          (const uint4*)(embh + (size_t)((jt + 1) * 16 + srow) * DIM + schunk * 16);
      uint4* dst = (uint4*)&Bt[buf ^ 1][srow][schunk * 16];
      dst[0] = src[0];
      dst[1] = src[1];
      if (jt + 2 < NT)
        __builtin_prefetch(embh + (size_t)((jt + 2) * 16 + srow) * DIM + schunk * 16, 0, 1);
    }

    const int ncol = jt * 16 + l15;               // column owned by this lane
    const int   colLab = labels[ncol];
    const int   colSbj = sbj[ncol];
    const float colSq  = sq[ncol];

    f32x8 c = {};
    const unsigned short* brow = &Bt[buf][l15][half * 8];
    #pragma unroll
    for (int kc = 0; kc < 8; ++kc) {
      const unsigned short* p = brow + kc * 32;
      bf16x8 lo = *(const bf16x8*)p;
      bf16x8 hi = *(const bf16x8*)(p + 16);
      bf16x16 b;
      #pragma unroll
      for (int e = 0; e < 8; ++e) { b[e] = lo[e]; b[e + 8] = hi[e]; }
      c = __builtin_amdgcn_wmma_f32_16x16x32_bf16(
              false, a[kc], false, b, (short)0, c, false, false);
    }

    // Epilogue: Gram -> distance -> masked running argmax/argmin
    #pragma unroll
    for (int r = 0; r < 8; ++r) {
      int   mrow = row0 + half * 8 + r;
      float d2   = mySq[r] + colSq - 2.0f * c[r];
      float dist = __builtin_sqrtf(fmaxf(d2, 0.0f));
      bool sameS = (mySbj[r] == colSbj);
      bool sameL = (myLab[r] == colLab);
      bool isPos = sameS && sameL && (mrow != ncol);
      bool isNeg = sameS && !sameL;
      if (isPos && dist > bestP[r]) { bestP[r] = dist; bpi[r] = ncol; }
      if (isNeg && dist < bestN[r]) { bestN[r] = dist; bni[r] = ncol; }
    }
  }

  // ---- Butterfly reduce across the 16 lanes sharing each row group ----
  #pragma unroll
  for (int m = 8; m >= 1; m >>= 1) {
    #pragma unroll
    for (int r = 0; r < 8; ++r) {
      float oP  = __shfl_xor(bestP[r], m);
      int   oPi = __shfl_xor(bpi[r],   m);
      if (oP > bestP[r]) { bestP[r] = oP; bpi[r] = oPi; }
      float oN  = __shfl_xor(bestN[r], m);
      int   oNi = __shfl_xor(bni[r],   m);
      if (oN < bestN[r]) { bestN[r] = oN; bni[r] = oNi; }
    }
  }

  if (l15 == 0) {
    #pragma unroll
    for (int r = 0; r < 8; ++r) {
      int mrow = row0 + half * 8 + r;
      hp[mrow] = bpi[r];
      hn[mrow] = bni[r];
    }
  }
}

// ---------------------------------------------------------------------------
// Kernel 3: fp32 recomputation of gathered distances + hinge + reduction.
// One wave per row; lanes stride dims.
// ---------------------------------------------------------------------------
__global__ __launch_bounds__(256) void WST_loss_kernel(
    const float* __restrict__ emb,
    const int* __restrict__ hp,
    const int* __restrict__ hn,
    float* __restrict__ sumcnt) {
  int i    = (int)((blockIdx.x * blockDim.x + threadIdx.x) >> 5);
  int lane = (int)(threadIdx.x & 31);
  if (i >= NROWS) return;

  int p = hp[i], n = hn[i];
  bool valid = (p >= 0) && (n >= 0);
  float sap = 0.0f, san = 0.0f;
  if (valid) {
    const float* ei = emb + (size_t)i * DIM;
    const float* ep = emb + (size_t)p * DIM;
    const float* en = emb + (size_t)n * DIM;
    #pragma unroll 4
    for (int d = lane; d < DIM; d += 32) {
      float x = ei[d];
      float dp = x - ep[d] + EPSV;  sap += dp * dp;
      float dn = x - en[d] + EPSV;  san += dn * dn;
    }
  }
  #pragma unroll
  for (int m = 16; m >= 1; m >>= 1) {
    sap += __shfl_xor(sap, m);
    san += __shfl_xor(san, m);
  }
  if (lane == 0 && valid) {
    float per = fmaxf(__builtin_sqrtf(sap) - __builtin_sqrtf(san) + MARGIN, 0.0f);
    atomicAdd(&sumcnt[0], per);
    atomicAdd(&sumcnt[1], 1.0f);
  }
}

// ---------------------------------------------------------------------------
// Kernel 4: final scalar
// ---------------------------------------------------------------------------
__global__ void WST_fin_kernel(const float* __restrict__ sumcnt,
                               float* __restrict__ out) {
  out[0] = sumcnt[0] / fmaxf(sumcnt[1], 1.0f);
}

// ---------------------------------------------------------------------------
extern "C" void kernel_launch(void* const* d_in, const int* in_sizes, int n_in,
                              void* d_out, int out_size, void* d_ws, size_t ws_size,
                              hipStream_t stream) {
  (void)in_sizes; (void)n_in; (void)out_size; (void)ws_size;
  const float* emb    = (const float*)d_in[0];
  const int*   labels = (const int*)d_in[1];
  const int*   sbj    = (const int*)d_in[2];

  char* w = (char*)d_ws;
  float*          sumcnt = (float*)w;                                   // 256 B header
  unsigned short* embh   = (unsigned short*)(w + 256);                  // 4 MiB
  float*          sq     = (float*)(w + 256 + (size_t)NROWS * DIM * 2); // 32 KiB
  int*            hp     = (int*)(w + 256 + (size_t)NROWS * DIM * 2 + NROWS * 4);
  int*            hn     = hp + NROWS;

  WST_init_kernel<<<1, 32, 0, stream>>>(sumcnt);
  WST_prep_kernel<<<NROWS / 8, 256, 0, stream>>>(emb, embh, sq);        // 8 rows/block
  WST_mine_kernel<<<(NROWS / 16) / 8, 256, 0, stream>>>(embh, sq, labels, sbj, hp, hn);
  WST_loss_kernel<<<NROWS / 8, 256, 0, stream>>>(emb, hp, hn, sumcnt);
  WST_fin_kernel<<<1, 1, 0, stream>>>(sumcnt, (float*)d_out);
}
